// VectorQuantization_16604343566481
// MI455X (gfx1250) — matrix operand order repaired and checked by reading, not compiled, not run on hardware
//
#include <hip/hip_runtime.h>
#include <hip/hip_bf16.h>

// VQ-VAE vector quantization for MI455X (gfx1250, wave32, WMMA).
//
// score(t,k) = ||e_k||^2 - 2 z_t . e_k   (per-row ||z||^2 dropped: constant for argmin)
// Double-f16 split keeps argmin f32-exact: s=-2z=s_hi+s_lo, e=e_hi+e_lo,
// dot = s_hi*e_hi + s_lo*e_hi + s_hi*e_lo  (lo*lo term ~1e-11, dropped).
// All matmul work goes through v_wmma_f32_16x16x32_f16.
// B tiles are double-buffered in LDS; staged via GLOBAL_LOAD_ASYNC_TO_LDS_B128
// (ASYNCcnt) so the memory->LDS DMA overlaps the WMMAs.

typedef __attribute__((ext_vector_type(16))) _Float16 v16h;
typedef __attribute__((ext_vector_type(8)))  float    v8f;

#define T_TOKENS   32768
#define K_CODES    8192
#define D_DIM      64
#define NTILE_K    (K_CODES / 16)          // 512 code tiles
#define TILES_PER_WAVE 2
#define WAVES_PER_BLOCK 4
#define BLOCK_THREADS  (WAVES_PER_BLOCK * 32)
#define MAIN_BLOCKS (T_TOKENS / (16 * TILES_PER_WAVE * WAVES_PER_BLOCK))  // 256

#if defined(__has_builtin)
#  if __has_builtin(__builtin_amdgcn_global_load_async_to_lds_b128)
#    define HAVE_ASYNC_LDS 1
#  endif
#  if __has_builtin(__builtin_amdgcn_s_wait_asynccnt)
#    define HAVE_WAIT_ASYNC_BUILTIN 1
#  endif
#endif
#ifndef HAVE_ASYNC_LDS
#  define HAVE_ASYNC_LDS 0
#endif
#ifndef HAVE_WAIT_ASYNC_BUILTIN
#  define HAVE_WAIT_ASYNC_BUILTIN 0
#endif

#if HAVE_ASYNC_LDS
// Builtin prototype (from clang diagnostic): b128 variant takes
// (v4i addrspace(1)*, v4i addrspace(3)*, imm i32 offset, imm i32 cpol).
typedef int v4i_ __attribute__((vector_size(16)));
typedef __attribute__((address_space(1))) v4i_* async_gptr;
typedef __attribute__((address_space(3))) v4i_* async_lptr;
#endif

__device__ __forceinline__ void wait_async_lds() {
#if HAVE_ASYNC_LDS
#  if HAVE_WAIT_ASYNC_BUILTIN
  __builtin_amdgcn_s_wait_asynccnt(0);
#  else
  asm volatile("s_wait_asynccnt 0x0" ::: "memory");
#  endif
#endif
}

__device__ __forceinline__ v8f wmma_f16(v16h a, v16h b, v8f c) {
  // (neg_a, A, neg_b, B, c_mod, C, reuse_a, reuse_b)
  return __builtin_amdgcn_wmma_f32_16x16x32_f16(false, a, false, b, (short)0, c,
                                                false, false);
}

// ---------------- prep: pack -2*z into WMMA A layout (hi/lo f16) ----------------
// A (16x32 f16) per-lane layout (ISA 7.12.2): lane L holds row M=L%16,
// f16 slot j in K-chunk maps to K = (j&7) + ((j>>3)<<4) + ((L>>4)<<3).
__global__ void vq_prep_z(const float* __restrict__ z,
                          _Float16* __restrict__ a_hi,
                          _Float16* __restrict__ a_lo) {
  int id   = blockIdx.x * blockDim.x + threadIdx.x;  // 131072 threads
  int tile  = id >> 6;        // token tile 0..2047
  int r     = id & 63;
  int chunk = r >> 5;         // K-chunk (D dims 0..31 / 32..63)
  int lane  = r & 31;
  int tok   = tile * 16 + (lane & 15);
  long base = ((long)(tile * 2 + chunk) * 32 + lane) * 16;
  v16h h, l;
#pragma unroll
  for (int j = 0; j < 16; ++j) {
    int dim = chunk * 32 + (j & 7) + ((j >> 3) << 4) + ((lane >> 4) << 3);
    float s = -2.0f * z[(long)tok * D_DIM + dim];
    _Float16 hi = (_Float16)s;
    h[j] = hi;
    l[j] = (_Float16)(s - (float)hi);
  }
  *(v16h*)(a_hi + base) = h;
  *(v16h*)(a_lo + base) = l;
}

// ---------------- prep: pack weight^T into WMMA B layout (hi/lo f16) -----------
// B (32x16 f16) per-lane layout: lane L holds column N=L%16,
// f16 slot j maps to K = (j&7) + ((j>>3)<<4) + ((L>>4)<<3).
__global__ void vq_prep_w(const float* __restrict__ w,
                          _Float16* __restrict__ b_hi,
                          _Float16* __restrict__ b_lo) {
  int id   = blockIdx.x * blockDim.x + threadIdx.x;  // 32768 threads
  int kt    = id >> 6;        // code tile 0..511
  int r     = id & 63;
  int chunk = r >> 5;
  int lane  = r & 31;
  int code  = kt * 16 + (lane & 15);
  long base = ((long)(kt * 2 + chunk) * 32 + lane) * 16;
  v16h h, l;
#pragma unroll
  for (int j = 0; j < 16; ++j) {
    int dim = chunk * 32 + (j & 7) + ((j >> 3) << 4) + ((lane >> 4) << 3);
    float s = w[(long)code * D_DIM + dim];
    _Float16 hi = (_Float16)s;
    h[j] = hi;
    l[j] = (_Float16)(s - (float)hi);
  }
  *(v16h*)(b_hi + base) = h;
  *(v16h*)(b_lo + base) = l;
}

// ---------------- prep: ||e_k||^2 ----------------
__global__ void vq_prep_norms(const float* __restrict__ w,
                              float* __restrict__ norms) {
  int k = blockIdx.x * blockDim.x + threadIdx.x;  // 8192 threads
  float acc = 0.0f;
  const float4* row = (const float4*)(w + (long)k * D_DIM);
#pragma unroll
  for (int d = 0; d < D_DIM / 4; ++d) {
    float4 v = row[d];
    acc += v.x * v.x + v.y * v.y + v.z * v.z + v.w * v.w;
  }
  norms[k] = acc;
}

// ---------------- main: fused distance-matmul + argmin + gather + loss --------
__global__ void __launch_bounds__(BLOCK_THREADS)
vq_argmin_wmma(const _Float16* __restrict__ a_hi,
               const _Float16* __restrict__ a_lo,
               const _Float16* __restrict__ b_hi,
               const _Float16* __restrict__ b_lo,
               const float* __restrict__ norms,
               const float* __restrict__ z,
               const float* __restrict__ weight,
               float* __restrict__ out,
               float* __restrict__ partials) {
  // double-buffered B tile: [buf][hi c0, hi c1, lo c0, lo c1][lane]  8KB
  __shared__ v16h  smemB[2][4][32];
  __shared__ int   sIdx[WAVES_PER_BLOCK][TILES_PER_WAVE][16];
  __shared__ float sRed[BLOCK_THREADS];

  const int tid  = threadIdx.x;
  const int lane = tid & 31;
  const int wave = tid >> 5;

  // Stage one B tile (hi/lo, both K-chunks) into LDS buffer `buf`.
  // 128 threads x 32B. Async memory->LDS DMA (ASYNCcnt) when available.
  auto stage = [&](int kt, int buf) {
    int slot  = tid >> 5;              // 0..3
    int sl    = tid & 31;
    int chunk = slot & 1;
    const _Float16* src = (slot < 2) ? b_hi : b_lo;
    const _Float16* g = src + ((long)(kt * 2 + chunk) * 32 + sl) * 16;
#if HAVE_ASYNC_LDS
    // IOFFSET applies to both global and LDS address (ISA async pseudocode)
    __builtin_amdgcn_global_load_async_to_lds_b128(
        (async_gptr)g, (async_lptr)&smemB[buf][slot][sl], 0, 0);
    __builtin_amdgcn_global_load_async_to_lds_b128(
        (async_gptr)g, (async_lptr)&smemB[buf][slot][sl], 16, 0);
#else
    smemB[buf][slot][sl] = *(const v16h*)g;
#endif
  };

  // Load this wave's A operands (kept resident in VGPRs for the whole K loop).
  v16h aH[TILES_PER_WAVE][2], aL[TILES_PER_WAVE][2];
  const int tileBase = (blockIdx.x * WAVES_PER_BLOCK + wave) * TILES_PER_WAVE;
#pragma unroll
  for (int t = 0; t < TILES_PER_WAVE; ++t)
#pragma unroll
    for (int c = 0; c < 2; ++c) {
      long off = ((long)((tileBase + t) * 2 + c) * 32 + lane) * 16;
      aH[t][c] = *(const v16h*)(a_hi + off);
      aL[t][c] = *(const v16h*)(a_lo + off);
    }

  float bv[TILES_PER_WAVE][8];
  int   bi[TILES_PER_WAVE][8];
#pragma unroll
  for (int t = 0; t < TILES_PER_WAVE; ++t)
#pragma unroll
    for (int i = 0; i < 8; ++i) { bv[t][i] = 3.4e38f; bi[t][i] = 0; }

  // Prologue: stage tile 0 into buffer 0.
  stage(0, 0);
  wait_async_lds();
  __syncthreads();

  for (int kt = 0; kt < NTILE_K; ++kt) {
    const int cur = kt & 1;
    // Kick off the DMA for the next tile immediately so it overlaps the WMMAs.
    if (kt + 1 < NTILE_K) stage(kt + 1, cur ^ 1);

    v16h bH0 = smemB[cur][0][lane];
    v16h bH1 = smemB[cur][1][lane];
    v16h bL0 = smemB[cur][2][lane];
    v16h bL1 = smemB[cur][3][lane];
    float nrm = norms[kt * 16 + (lane & 15)];
    int codeId = kt * 16 + (lane & 15);   // C layout: N = lane%16

#pragma unroll
    for (int t = 0; t < TILES_PER_WAVE; ++t) {
      v8f c;
#pragma unroll
      for (int i = 0; i < 8; ++i) c[i] = nrm;       // start from ||e||^2
      // dot = s_hi*e_hi + s_lo*e_hi + s_hi*e_lo  over both K-chunks
      c = wmma_f16(aH[t][0], bH0, c);
      c = wmma_f16(aH[t][1], bH1, c);
      c = wmma_f16(aL[t][0], bH0, c);
      c = wmma_f16(aL[t][1], bH1, c);
      c = wmma_f16(aH[t][0], bL0, c);
      c = wmma_f16(aH[t][1], bL1, c);
#pragma unroll
      for (int i = 0; i < 8; ++i) {
        if (c[i] < bv[t][i]) { bv[t][i] = c[i]; bi[t][i] = codeId; }
      }
    }

    // Next tile's DMA must have landed before any wave starts the next
    // iteration; the barrier also keeps waves from overwriting a buffer
    // that another wave is still reading.
    wait_async_lds();
    __syncthreads();
  }

  // Cross-lane argmin within each 16-lane column group (masks 1,2,4,8 stay
  // inside the group; lanes 0-15 and 16-31 hold different token rows).
#pragma unroll
  for (int t = 0; t < TILES_PER_WAVE; ++t) {
#pragma unroll
    for (int i = 0; i < 8; ++i) {
      float v = bv[t][i];
      int   ix = bi[t][i];
#pragma unroll
      for (int m = 1; m < 16; m <<= 1) {
        float ov = __shfl_xor(v, m, 32);
        int   oi = __shfl_xor(ix, m, 32);
        if (ov < v || (ov == v && oi < ix)) { v = ov; ix = oi; }
      }
      if ((lane & 15) == 0) {
        // C layout: VGPR i holds M=i (lanes 0-15) or M=i+8 (lanes 16-31)
        sIdx[wave][t][i + ((lane >> 4) << 3)] = ix;
      }
    }
  }
  __syncthreads();

  // Gather z_q = weight[idx] (f32), write straight-through output (== z_q),
  // accumulate per-block loss partial sum((z_e - z_q)^2).
  {
    int w2 = tid >> 5;
    int t2 = (tid >> 4) & 1;
    int m2 = tid & 15;
    long gtok = ((long)(blockIdx.x * WAVES_PER_BLOCK + w2) * TILES_PER_WAVE + t2) * 16 + m2;
    int  code = sIdx[w2][t2][m2];
    const float4* wrow = (const float4*)(weight + (long)code * D_DIM);
    const float4* zrow = (const float4*)(z + gtok * D_DIM);
    float4*       orow = (float4*)(out + gtok * D_DIM);
    float acc = 0.0f;
#pragma unroll
    for (int d = 0; d < D_DIM / 4; ++d) {
      float4 wq = wrow[d];
      float4 ze = zrow[d];
      orow[d] = wq;
      float dx = ze.x - wq.x, dy = ze.y - wq.y, dz = ze.z - wq.z, dw = ze.w - wq.w;
      acc += dx * dx + dy * dy + dz * dz + dw * dw;
    }
    sRed[tid] = acc;
  }
  __syncthreads();
  for (int off = BLOCK_THREADS / 2; off > 0; off >>= 1) {
    if (tid < off) sRed[tid] += sRed[tid + off];
    __syncthreads();
  }
  if (tid == 0) partials[blockIdx.x] = sRed[0];
}

// ---------------- finalize: deterministic fixed-order loss reduction ----------
__global__ void vq_finalize(const float* __restrict__ partials,
                            float* __restrict__ out) {
  __shared__ float s[MAIN_BLOCKS];
  s[threadIdx.x] = partials[threadIdx.x];
  __syncthreads();
  for (int off = MAIN_BLOCKS / 2; off > 0; off >>= 1) {
    if ((int)threadIdx.x < off) s[threadIdx.x] += s[threadIdx.x + off];
    __syncthreads();
  }
  if (threadIdx.x == 0) {
    float mse = s[0] / (float)(T_TOKENS * D_DIM);
    out[(long)T_TOKENS * D_DIM] = mse + 0.25f * mse;  // loss1 + 0.25*loss2
  }
}

extern "C" void kernel_launch(void* const* d_in, const int* in_sizes, int n_in,
                              void* d_out, int out_size, void* d_ws, size_t ws_size,
                              hipStream_t stream) {
  (void)in_sizes; (void)n_in; (void)out_size; (void)ws_size;
  const float* z      = (const float*)d_in[0];   // [32,1024,64] f32
  const float* weight = (const float*)d_in[1];   // [8192,64] f32
  float* out = (float*)d_out;                    // 2097152 z_q floats + 1 loss float

  char* ws = (char*)d_ws;
  _Float16* a_hi  = (_Float16*)(ws);
  _Float16* a_lo  = (_Float16*)(ws + (4u << 20));
  _Float16* b_hi  = (_Float16*)(ws + (8u << 20));
  _Float16* b_lo  = (_Float16*)(ws + (9u << 20));
  float* norms    = (float*)(ws + (10u << 20));
  float* partials = (float*)(ws + (10u << 20) + 32768);

  vq_prep_z<<<512, 256, 0, stream>>>(z, a_hi, a_lo);
  vq_prep_w<<<128, 256, 0, stream>>>(weight, b_hi, b_lo);
  vq_prep_norms<<<K_CODES / 256, 256, 0, stream>>>(weight, norms);
  vq_argmin_wmma<<<MAIN_BLOCKS, BLOCK_THREADS, 0, stream>>>(
      a_hi, a_lo, b_hi, b_lo, norms, z, weight, out, partials);
  vq_finalize<<<1, MAIN_BLOCKS, 0, stream>>>(partials, out);
}
